// MACELayer_29850022707544
// MI455X (gfx1250) — matrix operand order repaired and checked by reading, not compiled, not run on hardware
//
#include <hip/hip_runtime.h>
#include <hip/hip_bf16.h>

#define Nn   10000
#define En   320000
#define Cc   128
#define RBn  8
#define HIDn 64
#define NPATHn 5
#define EPSc 0.17677669529663687f

typedef __attribute__((ext_vector_type(16))) __bf16 v16bf;
typedef __attribute__((ext_vector_type(8)))  __bf16 v8bf;
typedef __attribute__((ext_vector_type(8)))  float  v8f;
typedef __attribute__((ext_vector_type(4)))  float  v4f;

__device__ __forceinline__ v8f wmma_bf16(v16bf a, v16bf b, v8f c) {
    // D = A(16x32 bf16) * B(32x16 bf16) + C(16x16 f32)
    return __builtin_amdgcn_wmma_f32_16x16x32_bf16(
        /*neg_a=*/false, a, /*neg_b=*/false, b,
        /*c_mod=*/(short)0, c, /*reuse_a=*/false, /*reuse_b=*/false);
}

// ---- fragment loaders (ISA 7.12.2 layouts) ----
// A 16x32 bf16: lane l: M=l%16, half=l/16; a[0..7] = K kb+half*8+0..7,
// a[8..15] = K kb+16+half*8+0..7  (two contiguous 8-elem runs)

// A from bf16 row-major (row stride ld, 16B-aligned rows): 2 x 16B loads
__device__ __forceinline__ v16bf load_a_frag_bf16(const __bf16* base, int ld, int kb) {
    int lane = threadIdx.x & 31;
    int m = lane & 15, half = lane >> 4;
    const __bf16* row = base + m * ld + kb + half * 8;
    v8bf lo = *(const v8bf*)(row);
    v8bf hi = *(const v8bf*)(row + 16);
    return __builtin_shufflevector(lo, hi, 0, 1, 2, 3, 4, 5, 6, 7,
                                   8, 9, 10, 11, 12, 13, 14, 15);
}

// B pre-packed in fragment order: one 32B load per lane
// packed index: (((tile_k * ntile_n) + tile_n) * 32 + lane) * 16 + j
__device__ __forceinline__ v16bf load_b_packed(const __bf16* pk, int tile_k, int tile_n,
                                               int ntile_n) {
    int lane = threadIdx.x & 31;
    const __bf16* p = pk + (((size_t)tile_k * ntile_n + tile_n) * 32 + lane) * 16;
    return *(const v16bf*)p;
}

// ---------------- prep kernels ----------------
// pack weight (K x Ncols, f32 row-major) into B-fragment order bf16
__global__ void pack_b_kernel(const float* __restrict__ src, __bf16* __restrict__ dst,
                              int K, int Ncols) {
    int idx = blockIdx.x * blockDim.x + threadIdx.x;
    int total = K * Ncols;
    if (idx >= total) return;
    int ntile_n = Ncols >> 4;
    int j    = idx & 15;
    int lane = (idx >> 4) & 31;
    int tile = idx >> 9;
    int tile_n = tile % ntile_n;
    int tile_k = tile / ntile_n;
    int half = lane >> 4;
    int k = tile_k * 32 + half * 8 + (j & 7) + ((j >> 3) << 4);     // b[i] holds K = half*8 + i (i<8), 16 + half*8 + (i-8) (i>=8)
    int n = tile_n * 16 + (lane & 15);
    dst[idx] = (__bf16)src[k * Ncols + n];
}

__global__ void zero_kernel(float* __restrict__ p, int n) {
    int i = blockIdx.x * blockDim.x + threadIdx.x;
    if (i < n) p[i] = 0.0f;
}

// ---------------- node up-projection: h_s = s@W_up_s, h_v[:,:,i] = v_i@W_up_v ----------------
__global__ __launch_bounds__(256)
void node_up_kernel(const float* __restrict__ node_feats,
                    const __bf16* __restrict__ pWup_s,
                    const __bf16* __restrict__ pWup_v,
                    float* __restrict__ h_s, float* __restrict__ h_v) {
    // bf16 staged planes: convert once, every wave reads bf16 directly
    __shared__ __align__(16) __bf16 s_lds[16][Cc];
    __shared__ __align__(16) __bf16 v_lds[3][16][Cc];
    int n0 = blockIdx.x * 16;
    int t = threadIdx.x;

    // coalesced stage of the whole 16x512 tile
    for (int idx = t; idx < 16 * 512; idx += 256) {
        int m = idx >> 9, q = idx & 511;
        float val = node_feats[(size_t)(n0 + m) * 512 + q];
        if (q < 128) s_lds[m][q] = (__bf16)val;
        else {
            int qq = q - 128;
            v_lds[qq % 3][m][qq / 3] = (__bf16)val;
        }
    }
    __syncthreads();

    int wave = t >> 5;
    int lane = t & 31;
    int half = lane >> 4, nidx = lane & 15;
#pragma unroll
    for (int comp = 0; comp < 4; ++comp) {
        const __bf16* W = (comp == 0) ? pWup_s : pWup_v;
        const __bf16* Abase = (comp == 0) ? &s_lds[0][0] : &v_lds[comp - 1][0][0];
        v8f acc = {};
#pragma unroll
        for (int kb = 0; kb < Cc; kb += 32) {
            acc = wmma_bf16(load_a_frag_bf16(Abase, Cc, kb),
                            load_b_packed(W, kb >> 5, wave, 8), acc);
        }
#pragma unroll
        for (int r = 0; r < 8; ++r) {
            int n = n0 + r + 8 * half;
            int d = wave * 16 + nidx;
            if (comp == 0) h_s[n * 128 + d] = acc[r];
            else           h_v[n * 384 + d * 3 + (comp - 1)] = acc[r];
        }
    }
}

// ---------------- fused edge kernel: radial MLP -> messages -> scatter atomics ----------------
__global__ __launch_bounds__(256)
void edge_kernel(const float* __restrict__ vectors,
                 const float* __restrict__ radial,
                 const int* __restrict__ senders,
                 const int* __restrict__ receivers,
                 const float* __restrict__ W_rad1,
                 const __bf16* __restrict__ pWrad2,
                 const float* __restrict__ h_s,
                 const float* __restrict__ h_v,
                 float* __restrict__ agg_s, float* __restrict__ agg_v) {
    __shared__ float  rad_lds[16][RBn];
    __shared__ __align__(16) __bf16 act_lds[16][HIDn];
    __shared__ float  y_lds[16][3];
    __shared__ int    snd_lds[16];
    __shared__ int    rcv_lds[16];

    int eb = blockIdx.x * 16;
    int t = threadIdx.x;

    if (t < 16) {
        int e = eb + t;
        float x = vectors[e * 3 + 0], y = vectors[e * 3 + 1], z = vectors[e * 3 + 2];
        float s3 = 1.7320508075688772f / (sqrtf(x * x + y * y + z * z) + 1e-9f);
        y_lds[t][0] = x * s3; y_lds[t][1] = y * s3; y_lds[t][2] = z * s3;
        snd_lds[t] = senders[e];
        rcv_lds[t] = receivers[e];
    }
    if (t < 128) {
        int el = t >> 3, r = t & 7;
        rad_lds[el][r] = radial[(eb + el) * RBn + r];
    }
    __syncthreads();

    // layer 1 (8->64) + SiLU, scalar (tiny)
    for (int idx = t; idx < 16 * HIDn; idx += 256) {
        int el = idx >> 6, h = idx & 63;
        float acc = 0.f;
#pragma unroll
        for (int r = 0; r < RBn; ++r) acc += rad_lds[el][r] * W_rad1[r * HIDn + h];
        float sig = 1.f / (1.f + __expf(-acc));
        act_lds[el][h] = (__bf16)(acc * sig);
    }
    __syncthreads();

    int wave = t >> 5, lane = t & 31;
    int cb = wave;                 // channel block 0..7
    int half = lane >> 4;
    int c = cb * 16 + (lane & 15); // this lane's channel

    // layer 2 (64 -> 5*128) via WMMA: A = act (16 edges x 64), reused across 5 paths
    v16bf a0 = load_a_frag_bf16(&act_lds[0][0], HIDn, 0);
    v16bf a1 = load_a_frag_bf16(&act_lds[0][0], HIDn, 32);

    v8f wfrag[NPATHn];
#pragma unroll
    for (int p = 0; p < NPATHn; ++p) {
        int tile_n = p * 8 + cb;   // col base (p*128 + cb*16) / 16, ntile_n = 40
        v8f acc = {};
        acc = wmma_bf16(a0, load_b_packed(pWrad2, 0, tile_n, 40), acc);
        acc = wmma_bf16(a1, load_b_packed(pWrad2, 1, tile_n, 40), acc);
        wfrag[p] = acc;
    }

    // messages + scatter (h_*/agg_* are L2-resident; fp32 atomics)
#pragma unroll
    for (int r = 0; r < 8; ++r) {
        int el = r + 8 * half;
        int sn = snd_lds[el], rn = rcv_lds[el];
        float y0 = y_lds[el][0], y1 = y_lds[el][1], y2 = y_lds[el][2];
        float sj  = h_s[sn * 128 + c];
        float vj0 = h_v[sn * 384 + c * 3 + 0];
        float vj1 = h_v[sn * 384 + c * 3 + 1];
        float vj2 = h_v[sn * 384 + c * 3 + 2];
        float dot = vj0 * y0 + vj1 * y1 + vj2 * y2;
        float cx = vj1 * y2 - vj2 * y1;
        float cy = vj2 * y0 - vj0 * y2;
        float cz = vj0 * y1 - vj1 * y0;
        float w0 = wfrag[0][r], w1 = wfrag[1][r], w2 = wfrag[2][r];
        float w3 = wfrag[3][r], w4 = wfrag[4][r];
        atomicAdd(&agg_s[rn * 128 + c], w0 * sj + w3 * dot);
        atomicAdd(&agg_v[rn * 384 + c * 3 + 0], w1 * sj * y0 + w2 * vj0 + w4 * cx);
        atomicAdd(&agg_v[rn * 384 + c * 3 + 1], w1 * sj * y1 + w2 * vj1 + w4 * cy);
        atomicAdd(&agg_v[rn * 384 + c * 3 + 2], w1 * sj * y2 + w2 * vj2 + w4 * cz);
    }
}

// ---------------- node finalize: down-proj, products, lin + skip, write new_feats ----------------
__global__ __launch_bounds__(256)
void node_final_kernel(const float* __restrict__ node_feats,
                       const int* __restrict__ node_specie,
                       const float* __restrict__ agg_s,
                       const float* __restrict__ agg_v,
                       const __bf16* __restrict__ pWdown_s,
                       const __bf16* __restrict__ pWdown_v,
                       const __bf16* __restrict__ pWlin_s,
                       const __bf16* __restrict__ pWlin_v,
                       const float* __restrict__ W_skip_s,
                       const float* __restrict__ W_skip_v,
                       const float* __restrict__ W_prod_s,
                       const float* __restrict__ W_prod_v,
                       float* __restrict__ out) {
    __shared__ __align__(16) __bf16 as_lds[16][Cc];        // agg_s, bf16
    __shared__ __align__(16) __bf16 ps_lds[16][Cc];        // p_s, bf16
    __shared__ __align__(16) __bf16 vplane[3][16][Cc];     // agg_v planes -> p_v planes

    int n0 = blockIdx.x * 16;
    int t = threadIdx.x, wave = t >> 5, lane = t & 31;
    int cb = wave, half = lane >> 4, nidx = lane & 15;
    int c = cb * 16 + nidx;

    // coalesced stage (convert once): agg_s + de-interleaved agg_v
    for (int idx = t; idx < 16 * 128; idx += 256) {
        int m = idx >> 7, q = idx & 127;
        as_lds[m][q] = (__bf16)agg_s[(size_t)(n0 + m) * 128 + q];
    }
    for (int idx = t; idx < 16 * 384; idx += 256) {
        int m = idx / 384, q = idx % 384;
        vplane[q % 3][m][q / 3] = (__bf16)agg_v[(size_t)(n0 + m) * 384 + q];
    }
    __syncthreads();

    // f_s = agg_s @ W_down_s * EPS
    v8f fs = {};
#pragma unroll
    for (int kb = 0; kb < Cc; kb += 32) {
        fs = wmma_bf16(load_a_frag_bf16(&as_lds[0][0], Cc, kb),
                       load_b_packed(pWdown_s, kb >> 5, cb, 8), fs);
    }
    // f_v[i] = agg_v[:,:,i] @ W_down_v * EPS
    v8f fv[3];
#pragma unroll
    for (int i = 0; i < 3; ++i) {
        v8f acc = {};
#pragma unroll
        for (int kb = 0; kb < Cc; kb += 32) {
            acc = wmma_bf16(load_a_frag_bf16(&vplane[i][0][0], Cc, kb),
                            load_b_packed(pWdown_v, kb >> 5, cb, 8), acc);
        }
        fv[i] = acc;
    }
    __syncthreads();   // all waves done reading vplane

    // per-specie tensor products -> p_s into ps_lds, p_v into vplane (reuse)
#pragma unroll
    for (int r = 0; r < 8; ++r) {
        int m = r + 8 * half;
        int n = n0 + m;
        int sp = node_specie[n];
        float f_s = fs[r] * EPSc;
        float f0 = fv[0][r] * EPSc, f1 = fv[1][r] * EPSc, f2 = fv[2][r] * EPSc;
        float vv = f0 * f0 + f1 * f1 + f2 * f2;
        const float* Wps = W_prod_s + sp * 3 * Cc;
        const float* Wpv = W_prod_v + sp * 2 * Cc;
        float p_s = f_s * Wps[c] + f_s * f_s * Wps[Cc + c] + vv * Wps[2 * Cc + c];
        float q0 = Wpv[c], q1 = Wpv[Cc + c];
        ps_lds[m][c] = (__bf16)p_s;
        vplane[0][m][c] = (__bf16)(f0 * q0 + f_s * f0 * q1);
        vplane[1][m][c] = (__bf16)(f1 * q0 + f_s * f1 * q1);
        vplane[2][m][c] = (__bf16)(f2 * q0 + f_s * f2 * q1);
    }
    __syncthreads();

    // out_s = p_s @ W_lin_s ; out_v[i] = p_v[:,:,i] @ W_lin_v
    v8f os = {};
#pragma unroll
    for (int kb = 0; kb < Cc; kb += 32) {
        os = wmma_bf16(load_a_frag_bf16(&ps_lds[0][0], Cc, kb),
                       load_b_packed(pWlin_s, kb >> 5, cb, 8), os);
    }
    v8f ov[3];
#pragma unroll
    for (int i = 0; i < 3; ++i) {
        v8f acc = {};
#pragma unroll
        for (int kb = 0; kb < Cc; kb += 32) {
            acc = wmma_bf16(load_a_frag_bf16(&vplane[i][0][0], Cc, kb),
                            load_b_packed(pWlin_v, kb >> 5, cb, 8), acc);
        }
        ov[i] = acc;
    }

    // add per-specie skip (scalar GEMV over c') and store new_feats
#pragma unroll
    for (int r = 0; r < 8; ++r) {
        int n = n0 + r + 8 * half;
        int d = c;
        int sp = node_specie[n];
        const float* Ws = W_skip_s + sp * Cc * Cc;
        const float* Wv = W_skip_v + sp * Cc * Cc;
        const float* nf_in = node_feats + (size_t)n * 512;
        float acc_s = os[r];
        float a0 = ov[0][r], a1 = ov[1][r], a2 = ov[2][r];
        for (int c2 = 0; c2 < Cc; ++c2) {
            float ws = Ws[c2 * Cc + d];
            float wv = Wv[c2 * Cc + d];
            acc_s += nf_in[c2] * ws;
            a0 += nf_in[128 + c2 * 3 + 0] * wv;
            a1 += nf_in[128 + c2 * 3 + 1] * wv;
            a2 += nf_in[128 + c2 * 3 + 2] * wv;
        }
        float* nf = out + Nn + (size_t)n * 512;
        nf[d] = acc_s;
        nf[128 + d * 3 + 0] = a0;
        nf[128 + d * 3 + 1] = a1;
        nf[128 + d * 3 + 2] = a2;
    }
}

// ---------------- readout: node_outputs = out_s @ W_ro ----------------
__global__ void readout_kernel(const float* __restrict__ W_ro, float* __restrict__ out) {
    int n = blockIdx.x * blockDim.x + threadIdx.x;
    if (n >= Nn) return;
    const float* os = out + Nn + (size_t)n * 512;
    float acc = 0.f;
#pragma unroll 8
    for (int cidx = 0; cidx < Cc; ++cidx) acc += os[cidx] * W_ro[cidx];
    out[n] = acc;
}

extern "C" void kernel_launch(void* const* d_in, const int* in_sizes, int n_in,
                              void* d_out, int out_size, void* d_ws, size_t ws_size,
                              hipStream_t stream) {
    const float* vectors     = (const float*)d_in[0];
    const float* node_feats  = (const float*)d_in[1];
    const int*   node_specie = (const int*)d_in[2];
    const float* radial      = (const float*)d_in[3];
    const int*   senders     = (const int*)d_in[4];
    const int*   receivers   = (const int*)d_in[5];
    const float* W_up_s      = (const float*)d_in[6];
    const float* W_up_v      = (const float*)d_in[7];
    const float* W_rad1      = (const float*)d_in[8];
    const float* W_rad2      = (const float*)d_in[9];
    const float* W_down_s    = (const float*)d_in[10];
    const float* W_down_v    = (const float*)d_in[11];
    const float* W_skip_s    = (const float*)d_in[12];
    const float* W_skip_v    = (const float*)d_in[13];
    const float* W_prod_s    = (const float*)d_in[14];
    const float* W_prod_v    = (const float*)d_in[15];
    const float* W_lin_s     = (const float*)d_in[16];
    const float* W_lin_v     = (const float*)d_in[17];
    const float* W_ro        = (const float*)d_in[18];
    float* out = (float*)d_out;

    // workspace carve-up (base is 256B aligned; all offsets keep >=32B alignment)
    char* ws = (char*)d_ws;
    float* h_s   = (float*)ws;                 ws += (size_t)Nn * 128 * 4;
    float* h_v   = (float*)ws;                 ws += (size_t)Nn * 384 * 4;
    float* agg_s = (float*)ws;                 ws += (size_t)Nn * 128 * 4;
    float* agg_v = (float*)ws;                 ws += (size_t)Nn * 384 * 4;
    __bf16* pWup_s   = (__bf16*)ws;            ws += 16384 * 2;
    __bf16* pWup_v   = (__bf16*)ws;            ws += 16384 * 2;
    __bf16* pWdown_s = (__bf16*)ws;            ws += 16384 * 2;
    __bf16* pWdown_v = (__bf16*)ws;            ws += 16384 * 2;
    __bf16* pWlin_s  = (__bf16*)ws;            ws += 16384 * 2;
    __bf16* pWlin_v  = (__bf16*)ws;            ws += 16384 * 2;
    __bf16* pWrad2   = (__bf16*)ws;            ws += 40960 * 2;

    const int CT = 256;
    pack_b_kernel<<<(16384 + CT - 1) / CT, CT, 0, stream>>>(W_up_s,   pWup_s,   Cc, Cc);
    pack_b_kernel<<<(16384 + CT - 1) / CT, CT, 0, stream>>>(W_up_v,   pWup_v,   Cc, Cc);
    pack_b_kernel<<<(16384 + CT - 1) / CT, CT, 0, stream>>>(W_down_s, pWdown_s, Cc, Cc);
    pack_b_kernel<<<(16384 + CT - 1) / CT, CT, 0, stream>>>(W_down_v, pWdown_v, Cc, Cc);
    pack_b_kernel<<<(16384 + CT - 1) / CT, CT, 0, stream>>>(W_lin_s,  pWlin_s,  Cc, Cc);
    pack_b_kernel<<<(16384 + CT - 1) / CT, CT, 0, stream>>>(W_lin_v,  pWlin_v,  Cc, Cc);
    pack_b_kernel<<<(40960 + CT - 1) / CT, CT, 0, stream>>>(W_rad2,   pWrad2,   HIDn, NPATHn * Cc);

    // agg_s and agg_v are contiguous: zero as one range (N*512 floats)
    zero_kernel<<<(Nn * 512 + CT - 1) / CT, CT, 0, stream>>>(agg_s, Nn * 512);

    node_up_kernel<<<Nn / 16, 256, 0, stream>>>(node_feats, pWup_s, pWup_v, h_s, h_v);

    edge_kernel<<<En / 16, 256, 0, stream>>>(vectors, radial, senders, receivers,
                                             W_rad1, pWrad2, h_s, h_v, agg_s, agg_v);

    node_final_kernel<<<Nn / 16, 256, 0, stream>>>(node_feats, node_specie, agg_s, agg_v,
                                                   pWdown_s, pWdown_v, pWlin_s, pWlin_v,
                                                   W_skip_s, W_skip_v, W_prod_s, W_prod_v,
                                                   out);

    readout_kernel<<<(Nn + CT - 1) / CT, CT, 0, stream>>>(W_ro, out);
}
